// RecurrentBaseline_40106404610784
// MI455X (gfx1250) — matrix operand order repaired and checked by reading, not compiled, not run on hardware
//
#include <hip/hip_runtime.h>
#include <hip/hip_bf16.h>

// ---------------------------------------------------------------------------
// RecurrentBaseline on MI455X (gfx1250): bf16-WMMA recurrent LSTM.
// Weights converted+swizzled to bf16 fragment layout once per call (118 MB,
// fits global L2); per-step GEMMs use v_wmma_f32_16x16x32_bf16 with f32
// accumulation; h/c/activations kept in f32. A-operands are staged into LDS
// with gfx1250 async LDS loads (ASYNCcnt) and double-buffered so staging
// overlaps the WMMA + B-operand L2 streaming.
// ---------------------------------------------------------------------------

typedef __bf16 bf16_t;
typedef bf16_t v16bf __attribute__((ext_vector_type(16)));
typedef float  v8f   __attribute__((ext_vector_type(8)));
typedef int    v4i   __attribute__((ext_vector_type(4)));

#define N_IN    4
#define N_HIDD  256
#define N_OUTD  4
#define N_ATOMS 10
#define HH      2560          // N_ATOMS * N_HID
#define G4      10240         // 4 * HH
#define BATCH   8
#define NSTEPS  99
#define KC_G    (HH / 32)     // 80 k-chunks of 32
#define NT_GATE (G4 / 16)     // 640 n-tiles
#define NT_F21  (HH / 16)     // 160 n-tiles
#define NT_F22  3             // ceil(40/16), cols 40..47 zero-padded
#define FRAG_ELEMS 512        // 32 lanes * 16 bf16 per fragment

// ---- gfx1250 async memory->LDS copy (ASYNCcnt), with safe fallback --------
#if defined(__has_builtin)
#  if __has_builtin(__builtin_amdgcn_global_load_async_to_lds_b128) && \
      __has_builtin(__builtin_amdgcn_s_wait_asynccnt)
#    define HAVE_ASYNC_LDS 1
#  endif
#endif
#ifndef HAVE_ASYNC_LDS
#  define HAVE_ASYNC_LDS 0
#endif

#if HAVE_ASYNC_LDS
// Signature (from hipcc diagnostic): (v4i AS1* gsrc, v4i AS3* lds_dst, Ii off, Ii cpol)
__device__ __forceinline__ void copy16_async(const bf16_t* g, bf16_t* l) {
    __builtin_amdgcn_global_load_async_to_lds_b128(
        (__attribute__((address_space(1))) v4i*)(g),
        (__attribute__((address_space(3))) v4i*)(l),
        /*offset=*/0, /*cpol=*/0);
}
#define ASYNC_FENCE() __builtin_amdgcn_s_wait_asynccnt(0)
#else
__device__ __forceinline__ void copy16_async(const bf16_t* g, bf16_t* l) {
    *(uint4*)l = *(const uint4*)g;      // sync fallback: b128 load + ds_store
}
#define ASYNC_FENCE() ((void)0)
#endif

// ---- A-fragment swizzled store: (row m, col k) -> [k/32][lane][e] ---------
// A 16x32 bf16 layout used here (same k order as the B swizzle): lane = m + 16*(k>=16 within chunk), e = k&15
__device__ __forceinline__ void store_A_swz(bf16_t* buf, int m, int k, float v) {
    int f    = k >> 5;
    int kk   = k & 31;
    int e    = kk & 15;
    int lane = m + ((kk >> 4) << 4);
    buf[(((size_t)f * 32 + lane) << 4) + e] = (bf16_t)v;
}

__device__ __forceinline__ float sigmoidf(float x) {
    return 1.0f / (1.0f + __expf(-x));
}

// ---------------------------------------------------------------------------
// One-time: convert fp32 weight (N x K, row-major, used as x @ W^T) into
// bf16 B-fragment layout: Bbuf[(tileN*Kchunks + kc)*32 + lane][e],
// where n = tileN*16 + lane%16 ; k = kc*32 + 16*(lane>=16) + e. Zero-pad OOB.
// ---------------------------------------------------------------------------
__global__ void swizzle_weight_kernel(const float* __restrict__ W, bf16_t* __restrict__ Bbuf,
                                      int N, int K, int Ntiles, int Kchunks) {
    int idx   = blockIdx.x * blockDim.x + threadIdx.x;
    int total = Ntiles * Kchunks * 32;
    if (idx >= total) return;
    int lane = idx & 31;
    int fk   = idx >> 5;
    int kc   = fk % Kchunks;
    int tn   = fk / Kchunks;
    int n     = tn * 16 + (lane & 15);
    int kbase = kc * 32 + ((lane >> 4) << 4);
    bf16_t* dst = Bbuf + (size_t)idx * 16;
#pragma unroll
    for (int e = 0; e < 16; ++e) {
        int k   = kbase + e;
        float v = (n < N && k < K) ? W[(size_t)n * K + k] : 0.0f;
        dst[e] = (bf16_t)v;
    }
}

__global__ void bias_sum_kernel(const float* __restrict__ bih, const float* __restrict__ bhh,
                                float* __restrict__ bs) {
    int i = blockIdx.x * blockDim.x + threadIdx.x;
    if (i < G4) bs[i] = bih[i] + bhh[i];
}

// ---------------------------------------------------------------------------
// Step kernel 1: ins = gt? input[t] : prev_out ; x1 = relu(fc1_1) ;
// x2 = relu(fc1_2) ; write x2 as swizzled bf16 A-fragments. 1 block per (b,a).
// ---------------------------------------------------------------------------
__global__ void __launch_bounds__(256)
fc1_kernel(const float* __restrict__ inputs, const float* __restrict__ prev_out,
           const int* __restrict__ ps, int t,
           const float* __restrict__ W1, const float* __restrict__ b1,
           const float* __restrict__ W2, const float* __restrict__ b2,
           bf16_t* __restrict__ Ax, float* __restrict__ ins_save) {
    __shared__ float ins_s[N_IN];
    __shared__ float x1_s[N_HIDD];
    int ba = blockIdx.x;
    int b  = ba / N_ATOMS, a = ba % N_ATOMS;
    int j  = threadIdx.x;
    bool use_gt = (t % ps[0]) == 0;
    if (j < N_IN) {
        float v = use_gt ? inputs[(((size_t)b * N_ATOMS + a) * 100 + t) * N_IN + j]
                         : prev_out[((size_t)b * N_ATOMS + a) * N_IN + j];
        ins_s[j] = v;
        ins_save[((size_t)b * N_ATOMS + a) * N_IN + j] = v;
    }
    __syncthreads();
    float acc = b1[j];
#pragma unroll
    for (int d = 0; d < N_IN; ++d) acc += ins_s[d] * W1[j * N_IN + d];
    x1_s[j] = fmaxf(acc, 0.0f);
    __syncthreads();
    float acc2 = b2[j];
    const float* w2row = W2 + (size_t)j * N_HIDD;
    for (int k = 0; k < N_HIDD; ++k) acc2 += x1_s[k] * w2row[k];
    float x2 = fmaxf(acc2, 0.0f);
    store_A_swz(Ax, b, a * N_HIDD + j, x2);
}

// ---------------------------------------------------------------------------
// Step kernel 2: gates = x@W_ih^T + h@W_hh^T + (b_ih+b_hh).
// 640 16-col tiles, 8 waves/block. A-fragments staged into LDS with async
// LDS loads, double-buffered in 8-chunk blocks (2 x 16 KB). 2 WMMAs/chunk.
// ---------------------------------------------------------------------------
#define KBLK_G 8
__global__ void __launch_bounds__(256)
gates_gemm_kernel(const bf16_t* __restrict__ Ax, const bf16_t* __restrict__ Ah,
                  const bf16_t* __restrict__ Bih, const bf16_t* __restrict__ Bhh,
                  const float* __restrict__ bias_sum, float* __restrict__ gates) {
    __shared__ __align__(128) bf16_t lds[2][2 * KBLK_G * FRAG_ELEMS];   // 2 x 16 KB
    const int lane = threadIdx.x & 31;
    const int wv   = threadIdx.x >> 5;
    const int tid  = threadIdx.x;
    const int tile = blockIdx.x * 8 + wv;                               // 0..639
    const size_t tileOff = (size_t)tile * KC_G * FRAG_ELEMS;
    v8f acc = {};

    auto stage = [&](int buf, int kb) {                                 // 16 KB total
        bf16_t* dx = &lds[buf][0];
        bf16_t* dh = &lds[buf][KBLK_G * FRAG_ELEMS];
        const bf16_t* sx = Ax + (size_t)kb * FRAG_ELEMS;
        const bf16_t* sh = Ah + (size_t)kb * FRAG_ELEMS;
#pragma unroll
        for (int i = 0; i < (KBLK_G * FRAG_ELEMS) / (256 * 8); ++i) {   // 2 iters
            int o = (tid + 256 * i) * 8;                                // 16B per copy
            copy16_async(sx + o, dx + o);
            copy16_async(sh + o, dh + o);
        }
    };

    stage(0, 0);
    ASYNC_FENCE();
    __syncthreads();
    int buf = 0;
    for (int kb = 0; kb < KC_G; kb += KBLK_G) {
        const int nkb = kb + KBLK_G;
        if (nkb < KC_G) stage(buf ^ 1, nkb);          // overlap fill with compute
        const bf16_t* lx = &lds[buf][0];
        const bf16_t* lh = &lds[buf][KBLK_G * FRAG_ELEMS];
#pragma unroll
        for (int kc = 0; kc < KBLK_G; ++kc) {
            v16bf ax = *(const v16bf*)(lx + kc * FRAG_ELEMS + lane * 16);
            v16bf ah = *(const v16bf*)(lh + kc * FRAG_ELEMS + lane * 16);
            v16bf bi = *(const v16bf*)(Bih + tileOff + (size_t)(kb + kc) * FRAG_ELEMS + lane * 16);
            v16bf bh = *(const v16bf*)(Bhh + tileOff + (size_t)(kb + kc) * FRAG_ELEMS + lane * 16);
            acc = __builtin_amdgcn_wmma_f32_16x16x32_bf16(false, ax, false, bi, (short)0, acc, false, false);
            acc = __builtin_amdgcn_wmma_f32_16x16x32_bf16(false, ah, false, bh, (short)0, acc, false, false);
        }
        if (nkb < KC_G) ASYNC_FENCE();
        __syncthreads();
        buf ^= 1;
    }
    if (lane < 16) {                                  // C/D: m=reg, n=lane
        int n = tile * 16 + lane;
        float bs = bias_sum[n];
#pragma unroll
        for (int r = 0; r < 8; ++r)
            gates[(size_t)r * G4 + n] = acc[r] + bs;
    }
}

// ---------------------------------------------------------------------------
// Step kernel 3: LSTM pointwise; c stays f32; h written as swizzled bf16 A.
// ---------------------------------------------------------------------------
__global__ void __launch_bounds__(256)
lstm_pointwise_kernel(const float* __restrict__ gates, float* __restrict__ c,
                      bf16_t* __restrict__ Ah) {
    int idx = blockIdx.x * blockDim.x + threadIdx.x;
    if (idx >= BATCH * HH) return;
    int m = idx / HH, n = idx % HH;
    const float* g = gates + (size_t)m * G4;
    float gi = g[n], gf = g[HH + n], gg = g[2 * HH + n], go = g[3 * HH + n];
    float cn = sigmoidf(gf) * c[idx] + sigmoidf(gi) * tanhf(gg);
    float hn = sigmoidf(go) * tanhf(cn);
    c[idx] = cn;
    store_A_swz(Ah, m, n, hn);
}

// ---------------------------------------------------------------------------
// Step kernel 4: y1 = relu(h @ fc2_1_w^T + b); 160 tiles; async double-
// buffered LDS staging (2 x 16 KB); output as swizzled bf16 A-fragments.
// ---------------------------------------------------------------------------
#define KBLK_F 16
__global__ void __launch_bounds__(256)
fc21_gemm_kernel(const bf16_t* __restrict__ Ah, const bf16_t* __restrict__ Bw,
                 const float* __restrict__ bias, bf16_t* __restrict__ Ay) {
    __shared__ __align__(128) bf16_t lds[2][KBLK_F * FRAG_ELEMS];       // 2 x 16 KB
    const int lane = threadIdx.x & 31;
    const int wv   = threadIdx.x >> 5;
    const int tid  = threadIdx.x;
    const int tile = blockIdx.x * 8 + wv;                               // 0..159
    const size_t tileOff = (size_t)tile * KC_G * FRAG_ELEMS;
    v8f acc = {};

    auto stage = [&](int buf, int kb) {                                 // 16 KB
        bf16_t* d = &lds[buf][0];
        const bf16_t* s = Ah + (size_t)kb * FRAG_ELEMS;
#pragma unroll
        for (int i = 0; i < (KBLK_F * FRAG_ELEMS) / (256 * 8); ++i) {   // 4 iters
            int o = (tid + 256 * i) * 8;
            copy16_async(s + o, d + o);
        }
    };

    stage(0, 0);
    ASYNC_FENCE();
    __syncthreads();
    int buf = 0;
    for (int kb = 0; kb < KC_G; kb += KBLK_F) {
        const int nkb = kb + KBLK_F;
        if (nkb < KC_G) stage(buf ^ 1, nkb);
        const bf16_t* la = &lds[buf][0];
#pragma unroll
        for (int kc = 0; kc < KBLK_F; ++kc) {
            v16bf a = *(const v16bf*)(la + kc * FRAG_ELEMS + lane * 16);
            v16bf b = *(const v16bf*)(Bw + tileOff + (size_t)(kb + kc) * FRAG_ELEMS + lane * 16);
            acc = __builtin_amdgcn_wmma_f32_16x16x32_bf16(false, a, false, b, (short)0, acc, false, false);
        }
        if (nkb < KC_G) ASYNC_FENCE();
        __syncthreads();
        buf ^= 1;
    }
    if (lane < 16) {
        int col = tile * 16 + lane;
        float bs = bias[col];
#pragma unroll
        for (int r = 0; r < 8; ++r)
            store_A_swz(Ay, r, col, fmaxf(acc[r] + bs, 0.0f));
    }
}

// ---------------------------------------------------------------------------
// Step kernel 5: y = y1 @ fc2_2_w^T + b + ins ; write output slice + prev_out.
// 3 tiles (N=40 padded to 48), one block of 3 waves.
// ---------------------------------------------------------------------------
__global__ void __launch_bounds__(96)
fc22_kernel(const bf16_t* __restrict__ Ay, const bf16_t* __restrict__ Bw,
            const float* __restrict__ bias, const float* __restrict__ ins_save,
            float* __restrict__ out, float* __restrict__ prev_out, int t) {
    const int lane = threadIdx.x & 31;
    const int tile = threadIdx.x >> 5;                            // 0..2
    const size_t tileOff = (size_t)tile * KC_G * FRAG_ELEMS;
    v8f acc = {};
    for (int kc = 0; kc < KC_G; ++kc) {
        v16bf a = *(const v16bf*)(Ay + (size_t)kc * FRAG_ELEMS + lane * 16);
        v16bf b = *(const v16bf*)(Bw + tileOff + (size_t)kc * FRAG_ELEMS + lane * 16);
        acc = __builtin_amdgcn_wmma_f32_16x16x32_bf16(false, a, false, b, (short)0, acc, false, false);
    }
    if (lane < 16) {
        int col = tile * 16 + lane;                               // 0..47
        if (col < N_ATOMS * N_OUTD) {
            float bs = bias[col];
            int a_ = col >> 2, o = col & 3;
#pragma unroll
            for (int r = 0; r < 8; ++r) {
                float v = acc[r] + bs + ins_save[r * (N_ATOMS * N_OUTD) + col];
                prev_out[r * (N_ATOMS * N_OUTD) + col] = v;
                out[((((size_t)r * N_ATOMS + a_) * NSTEPS) + t) * N_OUTD + o] = v;
            }
        }
    }
}

// ---------------------------------------------------------------------------
// Host side
// ---------------------------------------------------------------------------
extern "C" void kernel_launch(void* const* d_in, const int* in_sizes, int n_in,
                              void* d_out, int out_size, void* d_ws, size_t ws_size,
                              hipStream_t stream) {
    (void)in_sizes; (void)n_in; (void)out_size; (void)ws_size;
    const float* inputs  = (const float*)d_in[0];
    const float* fc11_w  = (const float*)d_in[1];
    const float* fc11_b  = (const float*)d_in[2];
    const float* fc12_w  = (const float*)d_in[3];
    const float* fc12_b  = (const float*)d_in[4];
    const float* w_ih    = (const float*)d_in[5];
    const float* w_hh    = (const float*)d_in[6];
    const float* b_ih    = (const float*)d_in[7];
    const float* b_hh    = (const float*)d_in[8];
    const float* fc21_w  = (const float*)d_in[9];
    const float* fc21_b  = (const float*)d_in[10];
    const float* fc22_w  = (const float*)d_in[11];
    const float* fc22_b  = (const float*)d_in[12];
    const int*   psteps  = (const int*)d_in[13];
    float* out = (float*)d_out;

    // Workspace carve (bytes), all offsets 256B-aligned. Total ~113.4 MB.
    char* ws = (char*)d_ws;
    const size_t szBih  = (size_t)NT_GATE * KC_G * FRAG_ELEMS * sizeof(bf16_t); // 52,428,800
    const size_t szB21  = (size_t)NT_F21  * KC_G * FRAG_ELEMS * sizeof(bf16_t); // 13,107,200
    const size_t szB22  = (size_t)NT_F22  * KC_G * FRAG_ELEMS * sizeof(bf16_t); //    245,760
    size_t off = 0;
    bf16_t* Bih   = (bf16_t*)(ws + off); off += szBih;
    bf16_t* Bhh   = (bf16_t*)(ws + off); off += szBih;
    bf16_t* B21   = (bf16_t*)(ws + off); off += szB21;
    bf16_t* B22   = (bf16_t*)(ws + off); off += szB22;
    float*  bsum  = (float*)(ws + off);  off += (size_t)G4 * sizeof(float);
    float*  gates = (float*)(ws + off);  off += (size_t)BATCH * G4 * sizeof(float);
    const size_t szA = (size_t)KC_G * FRAG_ELEMS * sizeof(bf16_t);              // 81,920
    char*   zero_base = ws + off;
    bf16_t* Ax    = (bf16_t*)(ws + off); off += szA;
    bf16_t* Ah    = (bf16_t*)(ws + off); off += szA;
    bf16_t* Ay    = (bf16_t*)(ws + off); off += szA;
    float*  cbuf  = (float*)(ws + off);  off += (size_t)BATCH * HH * sizeof(float);
    size_t  zero_bytes = (size_t)(ws + off - zero_base);   // Ax+Ah+Ay+c
    float*  ins_save = (float*)(ws + off); off += (size_t)BATCH * N_ATOMS * N_OUTD * sizeof(float);
    float*  prev_out = (float*)(ws + off); off += (size_t)BATCH * N_ATOMS * N_OUTD * sizeof(float);

    // Zero state + A-fragment padding rows (rows 8..15 stay zero forever).
    (void)hipMemsetAsync(zero_base, 0, zero_bytes, stream);

    // One-time weight conversion + swizzle into L2-resident bf16 fragments.
    {
        int tot = NT_GATE * KC_G * 32;
        swizzle_weight_kernel<<<(tot + 255) / 256, 256, 0, stream>>>(w_ih, Bih, G4, HH, NT_GATE, KC_G);
        swizzle_weight_kernel<<<(tot + 255) / 256, 256, 0, stream>>>(w_hh, Bhh, G4, HH, NT_GATE, KC_G);
        int t21 = NT_F21 * KC_G * 32;
        swizzle_weight_kernel<<<(t21 + 255) / 256, 256, 0, stream>>>(fc21_w, B21, HH, HH, NT_F21, KC_G);
        int t22 = NT_F22 * KC_G * 32;
        swizzle_weight_kernel<<<(t22 + 255) / 256, 256, 0, stream>>>(fc22_w, B22, N_ATOMS * N_OUTD, HH, NT_F22, KC_G);
        bias_sum_kernel<<<(G4 + 255) / 256, 256, 0, stream>>>(b_ih, b_hh, bsum);
    }

    // 99 recurrent steps; all weight reads hit L2 (118 MB bf16 < 192 MB).
    for (int t = 0; t < NSTEPS; ++t) {
        fc1_kernel<<<BATCH * N_ATOMS, 256, 0, stream>>>(inputs, prev_out, psteps, t,
                                                        fc11_w, fc11_b, fc12_w, fc12_b,
                                                        Ax, ins_save);
        gates_gemm_kernel<<<NT_GATE / 8, 256, 0, stream>>>(Ax, Ah, Bih, Bhh, bsum, gates);
        lstm_pointwise_kernel<<<(BATCH * HH + 255) / 256, 256, 0, stream>>>(gates, cbuf, Ah);
        fc21_gemm_kernel<<<NT_F21 / 8, 256, 0, stream>>>(Ah, B21, fc21_b, Ay);
        fc22_kernel<<<1, 96, 0, stream>>>(Ay, B22, fc22_b, ins_save, out, prev_out, t);
    }
}
